// NodeEmbedder_71296457114398
// MI455X (gfx1250) — matrix operand order repaired and checked by reading, not compiled, not run on hardware
//
#include <hip/hip_runtime.h>
#include <hip/hip_bf16.h>
#include <math.h>

// ---------------------------------------------------------------------------
// NodeEmbedder for MI455X (gfx1250, wave32, WMMA).
//
//  * prep_weights: W1..W4 (f32) -> bf16, zero-pad K, swizzled into WMMA
//    B-fragment tile order (16x32 tiles, 32B contiguous per lane) in d_ws.
//  * node_embed_kernel: workgroup (256 thr = 8 waves) owns 32 tokens:
//      - featurize into LDS as dense 32 x 1184 bf16
//      - 4 GEMM layers via v_wmma_f32_16x16x32_bf16; weight K-slices are
//        DOUBLE-BUFFERED into LDS with global_load_async_to_lds_b128
//        (ASYNCcnt), overlapping the global->LDS stream with WMMA compute;
//        activations live in LDS as bf16; f32 accumulate; bias+ReLU in
//        epilogues; final store masked by CA mask, f32 to d_out.
//
// ws usage: 368640 bf16 elements = 720 KB.
// LDS usage: ~140 KB (CDNA5 allows up to 320 KB per workgroup).
// ---------------------------------------------------------------------------

typedef float  v8f   __attribute__((ext_vector_type(8)));
typedef __bf16 v16bf __attribute__((ext_vector_type(16)));
typedef __bf16 v8bf  __attribute__((ext_vector_type(8)));

#define THREADS 256
#define MT      32            // tokens per workgroup
#define KPAD    1184          // 1157 padded to multiple of 32 (37 K-steps)
#define FSTRIDE 1192          // feature row stride (bank-conflict padding)
#define H1STR   264
#define H2STR   136
#define BBUF    8192          // bf16 elements per staging buffer (16 KB)

// ws element offsets (bf16 units)
#define W1S_OFF 0             // 37*16*512 = 303104
#define W2S_OFF 303104        // 8*8*512   = 32768
#define W3S_OFF 335872        // 4*8*512   = 16384
#define W4S_OFF 352256        // 4*8*512   = 16384

struct f3 { float x, y, z; };
__device__ __forceinline__ f3 f3sub(f3 a, f3 b) { return {a.x-b.x, a.y-b.y, a.z-b.z}; }
__device__ __forceinline__ f3 f3cross(f3 a, f3 b) {
  return {a.y*b.z - a.z*b.y, a.z*b.x - a.x*b.z, a.x*b.y - a.y*b.x};
}
__device__ __forceinline__ float f3dot(f3 a, f3 b) { return a.x*b.x + a.y*b.y + a.z*b.z; }
__device__ __forceinline__ f3 f3scale(f3 a, float s) { return {a.x*s, a.y*s, a.z*s}; }
__device__ __forceinline__ f3 f3norm(f3 a) {            // reference _normalize: /(|v|+eps)
  float n = sqrtf(f3dot(a, a)) + 1e-6f;
  return {a.x/n, a.y/n, a.z/n};
}
__device__ __forceinline__ f3 ld3(const float* p) { return {p[0], p[1], p[2]}; }

// dihedral with jnp.nan_to_num semantics
__device__ __forceinline__ float dihedral(f3 p0, f3 p1, f3 p2, f3 p3) {
  f3 v0 = f3sub(p2, p1), v1 = f3sub(p0, p1), v2 = f3sub(p3, p2);
  f3 u1 = f3cross(v0, v1), u2 = f3cross(v0, v2);
  float n1 = sqrtf(f3dot(u1, u1)), n2 = sqrtf(f3dot(u2, u2));
  float c = f3dot(u1, u2) / (n1 * n2);
  c = fminf(1.f, fmaxf(-1.f, c));
  float s  = f3dot(f3cross(v1, v2), v0);
  float sg = (float)(s > 0.f) - (float)(s < 0.f);
  float d  = sg * acosf(c);
  return (d == d) ? d : 0.f;
}

// ---------------------------------------------------------------------------
// Weight prep: f32 [Ksrc x N] row-major -> bf16 B-fragment-swizzled tiles.
// Tile (kt,nt) covers K=[kt*32,+32), N=[nt*16,+16), stored as 512 bf16:
//   lane = (n%16) + 16*((k%32)/16), pos = k%16, offset = lane*16 + pos
// so a wave's B fragment is 32B contiguous per lane.
// ---------------------------------------------------------------------------
__global__ void prep_weights(const float* __restrict__ W, __bf16* __restrict__ dst,
                             int Ksrc, int Kpad, int N) {
  int e = blockIdx.x * blockDim.x + threadIdx.x;
  int total = Kpad * N;
  if (e >= total) return;
  int k = e / N, n = e - k * N;
  float v = (k < Ksrc) ? W[k * N + n] : 0.f;
  int kt = k >> 5, nt = n >> 4;
  int lane = (n & 15) + (k & 16);
  int pos  = k & 15;
  int Ntiles = N >> 4;
  dst[(((size_t)kt * Ntiles + nt) << 9) + (lane << 4) + pos] = (__bf16)v;
}

// ---------------------------------------------------------------------------
// Async global->LDS staging of one K-slice (CHUNKS * 256 * 16B).
// Uses the CDNA5 direct-to-LDS path (ASYNCcnt), no VGPR round trip.
// ---------------------------------------------------------------------------
template <int CHUNKS>
__device__ __forceinline__ void stage_async(const __bf16* __restrict__ src,
                                            __bf16* dstLds, int tid) {
#pragma unroll
  for (int c = 0; c < CHUNKS; ++c) {
    int off = (tid + c * THREADS) * 8;                 // 16B per thread per chunk
    unsigned lds = (unsigned)(unsigned long long)(dstLds + off);
    unsigned long long ga = (unsigned long long)(src + off);
    asm volatile("global_load_async_to_lds_b128 %0, %1, off"
                 :: "v"(lds), "v"(ga) : "memory");
  }
}
__device__ __forceinline__ void wait_async0() {
  asm volatile("s_wait_asynccnt 0" ::: "memory");
}

// ---------------------------------------------------------------------------
// K-loop over one layer with double-buffered async weight staging.
// A frag layout (matches B swizzle): lane<16 holds row m=lane,
// K {0..15}; lane>=16 holds row m=lane-16, K {16..31}.
// ---------------------------------------------------------------------------
template <int NT, int CHUNKS>
__device__ __forceinline__ void mma_tiles(const __bf16* __restrict__ Abase, int strideA,
                                          const __bf16* __restrict__ gW, int ksteps,
                                          int nt0, v8f* acc,
                                          __bf16* sB, int tid, int lane) {
  const int slice = CHUNKS * THREADS * 8;              // elements per K-slice
  const __bf16* arow = Abase + (lane & 15) * strideA;

  __syncthreads();           // prior layer's compute / epilogue fully done
  stage_async<CHUNKS>(gW, sB, tid);                    // prologue: slice 0 -> buf 0

  for (int kt = 0; kt < ksteps; ++kt) {
    wait_async0();           // own async loads (into buf kt&1) complete
    __syncthreads();         // everyone's loads complete; prev compute done
    if (kt + 1 < ksteps)     // stream next slice into the other buffer
      stage_async<CHUNKS>(gW + (size_t)(kt + 1) * slice, sB + ((kt + 1) & 1) * BBUF, tid);

    const __bf16* cur = sB + (kt & 1) * BBUF;
    int kb = kt * 32 + ((lane & 16) ? 16 : 0);
    union { v16bf v; v8bf h[2]; } ua;
    ua.h[0] = *(const v8bf*)(arow + kb);
    ua.h[1] = *(const v8bf*)(arow + kb + 8);
#pragma unroll
    for (int t = 0; t < NT; ++t) {
      const __bf16* bp = cur + (nt0 + t) * 512 + lane * 16;
      union { v16bf v; v8bf h[2]; } ub;
      ub.h[0] = *(const v8bf*)bp;
      ub.h[1] = *(const v8bf*)(bp + 8);
      acc[t] = __builtin_amdgcn_wmma_f32_16x16x32_bf16(
          false, ua.v, false, ub.v, (short)0, acc[t], false, false);
    }
  }
}

// ---------------------------------------------------------------------------
// Fused featurize + 4-layer MLP. One workgroup = 32 tokens.
// ---------------------------------------------------------------------------
__global__ void __launch_bounds__(THREADS)
node_embed_kernel(const int* __restrict__ aa, const int* __restrict__ res_nb,
                  const int* __restrict__ chain_nb, const float* __restrict__ pos,
                  const int* __restrict__ mask_atoms, const int* __restrict__ smaskp,
                  const int* __restrict__ qmaskp, const float* __restrict__ emb,
                  const float* __restrict__ b1, const float* __restrict__ b2,
                  const float* __restrict__ b3, const float* __restrict__ b4,
                  const __bf16* __restrict__ W1s, const __bf16* __restrict__ W2s,
                  const __bf16* __restrict__ W3s, const __bf16* __restrict__ W4s,
                  float* __restrict__ out) {
  __shared__ __align__(16) __bf16 s_feat[MT * FSTRIDE];  // 76288 B
  __shared__ __align__(16) __bf16 s_b[2 * BBUF];         // 32768 B (dbl buffer)
  __shared__ __align__(16) __bf16 s_h1[MT * H1STR];      // 16896 B
  __shared__ __align__(16) __bf16 s_h2[MT * H2STR];      //  8704 B
  __shared__ __align__(16) __bf16 s_h3[MT * H2STR];      //  8704 B
  __shared__ float s_mres[MT];

  const int tid  = threadIdx.x;
  const int lane = tid & 31;
  const int wave = tid >> 5;
  const int tok0 = blockIdx.x * MT;

  // ---- phase 0: zero feature rows (16B chunks, 8 threads per token) ----
  {
    int t = tid >> 3, r = tid & 7;
    uint4 z = {0u, 0u, 0u, 0u};
    uint4* rowq = (uint4*)(s_feat + t * FSTRIDE);
    for (int c = r; c < (FSTRIDE * 2) / 16; c += 8) rowq[c] = z;
  }
  __syncthreads();

  // ---- phase 1: featurize 32 tokens ----
  {
    int t = tid >> 3, r = tid & 7;
    int g = tok0 + t, bI = g >> 10, l = g & 1023;
    int am = qmaskp[g] ? aa[g] : 21;
    if (am < 0 || am > 21) am = 21;
    __bf16* row = s_feat + t * FSTRIDE;
    const float* er = emb + am * 128;
    for (int j = r * 16; j < r * 16 + 16; ++j) row[j] = (__bf16)er[j];

    if (r == 0) {
      const float* P = pos + (size_t)g * 45;          // 15 atoms x 3
      f3 Np = ld3(P + 0), CAp = ld3(P + 3), Cp = ld3(P + 6);
      // basis R = [e1 e2 e3] from (center=CA, p1=C, p2=N)
      f3 e1 = f3norm(f3sub(Cp, CAp));
      f3 v2 = f3sub(Np, CAp);
      f3 e2 = f3norm(f3sub(v2, f3scale(e1, f3dot(e1, v2))));
      f3 e3 = f3cross(e1, e2);
      float sm = smaskp[g] ? 1.f : 0.f;
      int base = 128 + am * 45;
      for (int ai = 0; ai < 15; ++ai) {
        f3 p = ld3(P + ai * 3);
        f3 rel = f3sub(p, CAp);
        float f = (mask_atoms[g * 15 + ai] ? 1.f : 0.f) * sm;
        row[base + ai * 3 + 0] = (__bf16)(f3dot(e1, rel) * f);
        row[base + ai * 3 + 1] = (__bf16)(f3dot(e2, rel) * f);
        row[base + ai * 3 + 2] = (__bf16)(f3dot(e3, rel) * f);
      }
      // dihedrals (omega,phi at i use i-1; psi uses i+1)
      bool mres = mask_atoms[g * 15 + 1] != 0;        // CA mask
      s_mres[t] = mres ? 1.f : 0.f;
      bool cprev = false, cnext = false;
      if (l > 0) {
        int gp = g - 1;
        int dr = res_nb[g] - res_nb[gp]; if (dr < 0) dr = -dr;
        cprev = (dr == 1) && (chain_nb[g] == chain_nb[gp]) &&
                (mask_atoms[gp * 15 + 1] != 0) && mres;
      }
      if (l < 1023) {
        int gn = g + 1;
        int dr = res_nb[gn] - res_nb[g]; if (dr < 0) dr = -dr;
        cnext = (dr == 1) && (chain_nb[gn] == chain_nb[g]) &&
                (mask_atoms[gn * 15 + 1] != 0) && mres;
      }
      float omega = 0.f, phi = 0.f, psi = 0.f;
      if (cprev) {
        const float* Pp = pos + (size_t)(g - 1) * 45;
        f3 CAm = ld3(Pp + 3), Cm = ld3(Pp + 6);
        omega = dihedral(CAm, Cm, Np, CAp);
        phi   = dihedral(Cm, Np, CAp, Cp);
      }
      if (cnext) {
        const float* Pn = pos + (size_t)(g + 1) * 45;
        f3 Nn = ld3(Pn + 0);
        psi = dihedral(Np, CAp, Cp, Nn);
      }
      // dihed_mask: structure_mask & roll(+1) & roll(-1) (with wrap along L)
      int lp = (l + 1023) & 1023, ln = (l + 1) & 1023;
      bool dm = (smaskp[g] != 0) && (smaskp[bI * 1024 + lp] != 0) &&
                (smaskp[bI * 1024 + ln] != 0);
      const float FR[6] = {1.f, 2.f, 3.f, 1.f, 0.5f, 1.f / 3.f};
      float vals[3] = {omega, phi, psi};
      bool  msk[3]  = {cprev, cprev, cnext};
      for (int d = 0; d < 3; ++d) {
        float f = (msk[d] && dm) ? 1.f : 0.f;
        float v = vals[d];
        int c0 = 1118 + d * 13;
        row[c0] = (__bf16)(v * f);
#pragma unroll
        for (int j = 0; j < 6; ++j) {
          row[c0 + 1 + j] = (__bf16)(__sinf(v * FR[j]) * f);
          row[c0 + 7 + j] = (__bf16)(__cosf(v * FR[j]) * f);
        }
      }
    }
  }
  __syncthreads();

  const int nl = lane & 15, hi = (lane >> 4) & 1;

  // ---- layer 1: [32 x 1184] x [1184 x 256] ----
  {
    int mt = wave & 1, ng = wave >> 1;                // 4 N-groups of 64 cols
    v8f acc[4];
#pragma unroll
    for (int t = 0; t < 4; ++t)
#pragma unroll
      for (int v = 0; v < 8; ++v) acc[t][v] = 0.f;
    mma_tiles<4, 4>(s_feat + mt * 16 * FSTRIDE, FSTRIDE, W1s, KPAD / 32, ng * 4,
                    acc, s_b, tid, lane);
#pragma unroll
    for (int t = 0; t < 4; ++t) {
      int n0 = ng * 64 + t * 16;
      float bv = b1[n0 + nl];
#pragma unroll
      for (int v = 0; v < 8; ++v)
        s_h1[(mt * 16 + v + 8 * hi) * H1STR + n0 + nl] = (__bf16)fmaxf(acc[t][v] + bv, 0.f);
    }
  }

  // ---- layer 2: [32 x 256] x [256 x 128] ----
  {
    int mt = wave & 1, nt0 = (wave >> 1) * 2;
    v8f acc[2];
#pragma unroll
    for (int t = 0; t < 2; ++t)
#pragma unroll
      for (int v = 0; v < 8; ++v) acc[t][v] = 0.f;
    mma_tiles<2, 2>(s_h1 + mt * 16 * H1STR, H1STR, W2s, 8, nt0, acc, s_b, tid, lane);
#pragma unroll
    for (int t = 0; t < 2; ++t) {
      int n0 = (nt0 + t) * 16;
      float bv = b2[n0 + nl];
#pragma unroll
      for (int v = 0; v < 8; ++v)
        s_h2[(mt * 16 + v + 8 * hi) * H2STR + n0 + nl] = (__bf16)fmaxf(acc[t][v] + bv, 0.f);
    }
  }

  // ---- layer 3: [32 x 128] x [128 x 128] ----
  {
    int mt = wave & 1, nt0 = (wave >> 1) * 2;
    v8f acc[2];
#pragma unroll
    for (int t = 0; t < 2; ++t)
#pragma unroll
      for (int v = 0; v < 8; ++v) acc[t][v] = 0.f;
    mma_tiles<2, 2>(s_h2 + mt * 16 * H2STR, H2STR, W3s, 4, nt0, acc, s_b, tid, lane);
#pragma unroll
    for (int t = 0; t < 2; ++t) {
      int n0 = (nt0 + t) * 16;
      float bv = b3[n0 + nl];
#pragma unroll
      for (int v = 0; v < 8; ++v)
        s_h3[(mt * 16 + v + 8 * hi) * H2STR + n0 + nl] = (__bf16)fmaxf(acc[t][v] + bv, 0.f);
    }
  }

  // ---- layer 4: [32 x 128] x [128 x 128], + b4, * CA-mask, f32 store ----
  {
    int mt = wave & 1, nt0 = (wave >> 1) * 2;
    v8f acc[2];
#pragma unroll
    for (int t = 0; t < 2; ++t)
#pragma unroll
      for (int v = 0; v < 8; ++v) acc[t][v] = 0.f;
    mma_tiles<2, 2>(s_h3 + mt * 16 * H2STR, H2STR, W4s, 4, nt0, acc, s_b, tid, lane);
#pragma unroll
    for (int t = 0; t < 2; ++t) {
      int n0 = (nt0 + t) * 16;
      float bv = b4[n0 + nl];
#pragma unroll
      for (int v = 0; v < 8; ++v) {
        int m = mt * 16 + v + 8 * hi;
        out[(size_t)(tok0 + m) * 128 + n0 + nl] = (acc[t][v] + bv) * s_mres[m];
      }
    }
  }
}

// ---------------------------------------------------------------------------
extern "C" void kernel_launch(void* const* d_in, const int* in_sizes, int n_in,
                              void* d_out, int out_size, void* d_ws, size_t ws_size,
                              hipStream_t stream) {
  const int*   aa         = (const int*)d_in[0];
  const int*   res_nb     = (const int*)d_in[1];
  const int*   chain_nb   = (const int*)d_in[2];
  const float* pos        = (const float*)d_in[3];
  const int*   mask_atoms = (const int*)d_in[4];   // bool input, treated as i32
  const int*   smask      = (const int*)d_in[5];
  const int*   qmask      = (const int*)d_in[6];
  const float* emb        = (const float*)d_in[7];
  const float* W1 = (const float*)d_in[8];
  const float* b1 = (const float*)d_in[9];
  const float* W2 = (const float*)d_in[10];
  const float* b2 = (const float*)d_in[11];
  const float* W3 = (const float*)d_in[12];
  const float* b3 = (const float*)d_in[13];
  const float* W4 = (const float*)d_in[14];
  const float* b4 = (const float*)d_in[15];
  float* out = (float*)d_out;

  __bf16* wsb = (__bf16*)d_ws;                      // needs ~720 KB of ws
  __bf16* W1s = wsb + W1S_OFF;
  __bf16* W2s = wsb + W2S_OFF;
  __bf16* W3s = wsb + W3S_OFF;
  __bf16* W4s = wsb + W4S_OFF;

  prep_weights<<<(KPAD * 256 + 255) / 256, 256, 0, stream>>>(W1, W1s, 1157, KPAD, 256);
  prep_weights<<<(256 * 128 + 255) / 256, 256, 0, stream>>>(W2, W2s, 256, 256, 128);
  prep_weights<<<(128 * 128 + 255) / 256, 256, 0, stream>>>(W3, W3s, 128, 128, 128);
  prep_weights<<<(128 * 128 + 255) / 256, 256, 0, stream>>>(W4, W4s, 128, 128, 128);

  node_embed_kernel<<<(32 * 1024) / MT, THREADS, 0, stream>>>(
      aa, res_nb, chain_nb, pos, mask_atoms, smask, qmask, emb,
      b1, b2, b3, b4, W1s, W2s, W3s, W4s, out);
}
